// SpatialLoss_4724464025602
// MI455X (gfx1250) — matrix operand order repaired and checked by reading, not compiled, not run on hardware
//
#include <hip/hip_runtime.h>

typedef __attribute__((ext_vector_type(8)))  float          v8f;
typedef __attribute__((ext_vector_type(16))) __bf16         v16bf;
typedef __attribute__((ext_vector_type(2)))  __bf16         v2bf;
typedef __attribute__((ext_vector_type(4)))  unsigned int   v4u;
typedef __attribute__((ext_vector_type(8)))  unsigned int   v8u;

constexpr int CB = 64;    // batch
constexpr int CC = 768;   // channels
constexpr int HW = 576;   // 24*24 spatial positions
constexpr int NM = 50;    // num matches
constexpr int LSTR = 40;  // LDS row stride in half-words (80B, 16B-aligned)
constexpr int NSL = 16;   // row slices for filt moments
constexpr int CZ = 4;     // split-K slices for big cov
constexpr int TT = 192;   // GEMM tile edge (12 subtiles of 16)
constexpr int NW = 12;    // waves per GEMM block
constexpr int NTH = NW * 32;

// ---------------------------------------------------------------- helpers
static __device__ __forceinline__ unsigned pk2bf(float a, float b) {
  v2bf p;
  p[0] = static_cast<__bf16>(a);
  p[1] = static_cast<__bf16>(b);
  return __builtin_bit_cast(unsigned, p);
}

static __device__ __forceinline__ v16bf mkfrag(v4u lo, v4u hi) {
  v8u u;
#pragma unroll
  for (int i = 0; i < 4; ++i) { u[i] = lo[i]; u[i + 4] = hi[i]; }
  return __builtin_bit_cast(v16bf, u);
}

static __device__ __forceinline__ unsigned long long umin64(unsigned long long a,
                                                            unsigned long long b) {
  return a < b ? a : b;
}

// xor-butterfly lane exchange via ds_swizzle (group-of-32: and=0x1F, xor=M)
template <int M>
static __device__ __forceinline__ unsigned long long swz64(unsigned long long v) {
  constexpr int imm = (M << 10) | 0x1F;
  unsigned lo = (unsigned)v, hi = (unsigned)(v >> 32);
  lo = (unsigned)__builtin_amdgcn_ds_swizzle((int)lo, imm);
  hi = (unsigned)__builtin_amdgcn_ds_swizzle((int)hi, imm);
  return ((unsigned long long)hi << 32) | lo;
}

// ---------------------------------------------------------------- init
__global__ void vic_init(float* s, unsigned long long* nnmin) {
  int i = blockIdx.x * blockDim.x + threadIdx.x;
  if (blockIdx.x == 0 && threadIdx.x < 32) s[threadIdx.x] = 0.0f;
  if (i < 2 * CB * HW) nnmin[i] = ~0ULL;
}

// ---------------------------------------------------------------- pooled stats
__global__ void vic_pooled_stats(const float* __restrict__ P1, const float* __restrict__ P2,
                                 float* mean1, float* mean2, float* s) {
  int ch = blockIdx.x * blockDim.x + threadIdx.x;   // 0..767
  float m1 = 0.f, m2 = 0.f, sinv = 0.f;
  for (int b = 0; b < CB; ++b) {
    float a = P1[b * CC + ch], bb = P2[b * CC + ch];
    m1 += a; m2 += bb;
    float d = a - bb; sinv += d * d;
  }
  m1 *= (1.0f / CB); m2 *= (1.0f / CB);
  mean1[ch] = m1; mean2[ch] = m2;
  float v1 = 0.f, v2 = 0.f;
  for (int b = 0; b < CB; ++b) {
    float a = P1[b * CC + ch] - m1, bb = P2[b * CC + ch] - m2;
    v1 += a * a; v2 += bb * bb;
  }
  v1 *= (1.0f / (CB - 1)); v2 *= (1.0f / (CB - 1));
  atomicAdd(&s[0], sinv);
  atomicAdd(&s[1], fmaxf(1.0f - sqrtf(v1 + 1e-5f), 0.0f));
  atomicAdd(&s[2], fmaxf(1.0f - sqrtf(v2 + 1e-5f), 0.0f));
}

// ---------------------------------------------------------------- squared norms per map position
__global__ void vic_sqsum(const float* __restrict__ S1, const float* __restrict__ S2,
                          float* sq1, float* sq2) {
  int i = blockIdx.x * blockDim.x + threadIdx.x;   // 0 .. 64*576-1
  int b = i / HW, p = i % HW;
  float a1 = 0.f, a2 = 0.f;
  const float* p1 = S1 + (size_t)b * CC * HW + p;
  const float* p2 = S2 + (size_t)b * CC * HW + p;
  for (int c = 0; c < CC; ++c) {
    float x = p1[(size_t)c * HW]; a1 += x * x;
    float y = p2[(size_t)c * HW]; a2 += y * y;
  }
  sq1[i] = a1; sq2[i] = a2;
}

// ---------------------------------------------------------------- distance GEMM + NN match
// grid (3,3,64), block 384 (12 waves). 192x192 output tile per block, K=768 via
// bf16 WMMA 16x16x32; wave w owns m-subtile w x 12 n-subtiles (12 accumulators).
// NN ranking uses squared distance (sqrt is monotone; reference only uses
// distances for argmin/top-k). Packed (d2bits,idx) u64 atomic-min gives argmin
// with first-index tie-break for both directions in one pass (d21 = d12^T).
__global__ __launch_bounds__(NTH) void vic_dist(const float* __restrict__ S1,
                                                const float* __restrict__ S2,
                                                const float* __restrict__ sq1,
                                                const float* __restrict__ sq2,
                                                unsigned long long* nn1,
                                                unsigned long long* nn2) {
  const int b = blockIdx.z, m0 = blockIdx.x * TT, n0 = blockIdx.y * TT;
  __shared__ unsigned short As[TT][LSTR];
  __shared__ unsigned short Bs[TT][LSTR];
  const int tid = threadIdx.x, lane = tid & 31, w = tid >> 5;
  const float* A = S1 + (size_t)b * CC * HW;
  const float* Bp = S2 + (size_t)b * CC * HW;

  v8f acc[NW];
#pragma unroll
  for (int j = 0; j < NW; ++j)
#pragma unroll
    for (int r = 0; r < 8; ++r) acc[j][r] = 0.0f;

  const int lcol = lane & 15;
  const int mrow = 16 * w + lcol;
  const int klo = (lane < 16) ? 0 : 8;
  const int khalf = (lane < 16) ? 0 : 16;

  for (int kc = 0; kc < CC / 32; ++kc) {
#pragma unroll
    for (int g = 0; g < 4; ++g) {
      int gid = tid + g * NTH;            // 0..1535
      int kg = gid / TT;                  // 0..7 (4 k's each)
      int m = gid - kg * TT;              // 0..191
      const float* pa = A + (size_t)(kc * 32 + kg * 4) * HW + m0 + m;
      const float* pb = Bp + (size_t)(kc * 32 + kg * 4) * HW + n0 + m;
      unsigned a01 = pk2bf(pa[0], pa[(size_t)HW]);
      unsigned a23 = pk2bf(pa[(size_t)2 * HW], pa[(size_t)3 * HW]);
      unsigned b01 = pk2bf(pb[0], pb[(size_t)HW]);
      unsigned b23 = pk2bf(pb[(size_t)2 * HW], pb[(size_t)3 * HW]);
      *(unsigned long long*)&As[m][kg * 4] = a01 | ((unsigned long long)a23 << 32);
      *(unsigned long long*)&Bs[m][kg * 4] = b01 | ((unsigned long long)b23 << 32);
    }
    __syncthreads();

    v16bf a = mkfrag(*(const v4u*)&As[mrow][klo], *(const v4u*)&As[mrow][klo + 16]);
#pragma unroll
    for (int j = 0; j < NW; ++j) {
      int ncol = 16 * j + lcol;
      v16bf bb = mkfrag(*(const v4u*)&Bs[ncol][klo], *(const v4u*)&Bs[ncol][klo + 16]);
      acc[j] = __builtin_amdgcn_wmma_f32_16x16x32_bf16(
          false, a, false, bb, (short)0, acc[j], false, false);
    }
    __syncthreads();
  }

  // epilogue: squared distances + row/col argmin
  const int half8 = (lane < 16) ? 0 : 8;
  float b2[NW];
#pragma unroll
  for (int j = 0; j < NW; ++j) b2[j] = sq2[b * HW + n0 + 16 * j + lcol];
  unsigned long long colkey[NW];
#pragma unroll
  for (int j = 0; j < NW; ++j) colkey[j] = ~0ULL;
#pragma unroll
  for (int r = 0; r < 8; ++r) {
    int m = m0 + 16 * w + r + half8;
    float a2 = sq1[b * HW + m];
    unsigned long long rbest = ~0ULL;
#pragma unroll
    for (int j = 0; j < NW; ++j) {
      int n = n0 + 16 * j + lcol;
      float d2 = fmaxf(a2 + b2[j] - 2.0f * acc[j][r], 0.0f);
      unsigned long long db = (unsigned long long)__float_as_uint(d2) << 32;
      rbest = umin64(rbest, db | (unsigned)n);
      colkey[j] = umin64(colkey[j], db | (unsigned)m);
    }
    rbest = umin64(rbest, swz64<1>(rbest));
    rbest = umin64(rbest, swz64<2>(rbest));
    rbest = umin64(rbest, swz64<4>(rbest));
    rbest = umin64(rbest, swz64<8>(rbest));
    if (lcol == 0) atomicMin(&nn1[b * HW + m], rbest);
  }
#pragma unroll
  for (int j = 0; j < NW; ++j) {
    unsigned long long ck = umin64(colkey[j], swz64<16>(colkey[j]));
    if (lane < 16) atomicMin(&nn2[b * HW + n0 + 16 * j + lcol], ck);
  }
}

// ---------------------------------------------------------------- top-50 select + gather
__global__ __launch_bounds__(576) void vic_select(const float* __restrict__ S1,
                                                  const float* __restrict__ S2,
                                                  const unsigned long long* __restrict__ nnmin,
                                                  float* __restrict__ filt, float* s) {
  const int b = blockIdx.x, dir = blockIdx.y, t = threadIdx.x;
  __shared__ unsigned long long ksh[HW];
  __shared__ int sin[NM], scd[NM];
  const unsigned long long* nn = nnmin + ((size_t)dir * CB + b) * HW;
  ksh[t] = nn[t];
  __syncthreads();
  unsigned long long kt = ksh[t];
  float dt = __uint_as_float((unsigned)(kt >> 32));
  int rank = 0;
  for (int u = 0; u < HW; ++u) {
    float du = __uint_as_float((unsigned)(ksh[u] >> 32));
    rank += (du < dt) || (du == dt && u < t);
  }
  if (rank < NM) { sin[rank] = t; scd[rank] = (int)(unsigned)kt; }
  __syncthreads();

  const float* Sin = dir ? S2 : S1;
  const float* Scd = dir ? S1 : S2;
  float* fin = filt + (((size_t)(dir * 2 + 0) * CB + b) * NM) * CC;
  float* fcd = filt + (((size_t)(dir * 2 + 1) * CB + b) * NM) * CC;
  float invacc = 0.0f;
  for (int q = t; q < NM * CC; q += HW) {
    int j = q / CC, c = q % CC;
    float vi = Sin[((size_t)b * CC + c) * HW + sin[j]];
    float vc = Scd[((size_t)b * CC + c) * HW + scd[j]];
    fin[(size_t)j * CC + c] = vi;
    fcd[(size_t)j * CC + c] = vc;
    float d = vi - vc;
    invacc += d * d;
  }
#pragma unroll
  for (int off = 16; off; off >>= 1) invacc += __shfl_xor(invacc, off, 32);
  if ((t & 31) == 0) atomicAdd(&s[13 + dir], invacc);
}

// ---------------------------------------------------------------- matched-map moments
__global__ void vic_filt_moments(const float* __restrict__ filt,
                                 float* mpart, float* m2part) {
  const int q = blockIdx.y, sl = blockIdx.z;
  const int ch = blockIdx.x * blockDim.x + threadIdx.x;
  const int rows = (CB * NM) / NSL;   // 200
  const float* X = filt + (size_t)q * (CB * NM) * CC + (size_t)sl * rows * CC + ch;
  float s1 = 0.f, s2 = 0.f;
  for (int r = 0; r < rows; ++r) {
    float x = X[(size_t)r * CC];
    s1 += x; s2 += x * x;
  }
  mpart[((q * NSL) + sl) * CC + ch]  = s1;
  m2part[((q * NSL) + sl) * CC + ch] = s2;
}

// fixed-order slice reduction -> channel mean (for cov) + std term
__global__ void vic_filt_fin(const float* __restrict__ mpart,
                             const float* __restrict__ m2part,
                             float* meanf, float* s) {
  int idx = blockIdx.x * blockDim.x + threadIdx.x;   // 0..3071
  int q = idx / CC, ch = idx % CC;
  const int n = CB * NM;   // 3200
  float s1 = 0.f, s2 = 0.f;
  for (int sl = 0; sl < NSL; ++sl) {
    s1 += mpart[((q * NSL) + sl) * CC + ch];
    s2 += m2part[((q * NSL) + sl) * CC + ch];
  }
  float mean = s1 * (1.0f / n);
  meanf[q * CC + ch] = mean;
  float var = (s2 - (float)n * mean * mean) * (1.0f / (n - 1));
  atomicAdd(&s[5 + q], fmaxf(1.0f - sqrtf(var + 1e-5f), 0.0f));
}

// ---------------------------------------------------------------- covariance Gram, split-K
// grid (4,4,Z), block 384 (12 waves). 192x192 partial Gram tile over this
// block's K-slice, written to its private Gpart[z] region (no atomics).
__global__ __launch_bounds__(NTH) void vic_cov_part(const float* __restrict__ X,
                                                    const float* __restrict__ mean,
                                                    int n, float* __restrict__ Gpart) {
  const int i0 = blockIdx.x * TT, j0 = blockIdx.y * TT, z = blockIdx.z;
  __shared__ unsigned short As[TT][LSTR];
  __shared__ unsigned short Bs[TT][LSTR];
  const int tid = threadIdx.x, lane = tid & 31, w = tid >> 5;

  float mA[4], mB[4];
#pragma unroll
  for (int g = 0; g < 4; ++g) {
    int gid = tid + g * NTH;
    int m = gid - (gid / TT) * TT;
    mA[g] = mean[i0 + m];
    mB[g] = mean[j0 + m];
  }

  v8f acc8[NW];
#pragma unroll
  for (int j = 0; j < NW; ++j)
#pragma unroll
    for (int r = 0; r < 8; ++r) acc8[j][r] = 0.0f;

  const int lcol = lane & 15;
  const int mrow = 16 * w + lcol;
  const int klo = (lane < 16) ? 0 : 8;
  const int khalf = (lane < 16) ? 0 : 16;

  const int nchunk = n >> 5;
  const int cps = (nchunk + gridDim.z - 1) / gridDim.z;
  const int kcb = z * cps;
  const int kce = min(nchunk, kcb + cps);
  for (int kc = kcb; kc < kce; ++kc) {
#pragma unroll
    for (int g = 0; g < 4; ++g) {
      int gid = tid + g * NTH;
      int kg = gid / TT;
      int m = gid - kg * TT;
      const float* pa = X + (size_t)(kc * 32 + kg * 4) * CC + i0 + m;
      const float* pb = X + (size_t)(kc * 32 + kg * 4) * CC + j0 + m;
      unsigned a01 = pk2bf(pa[0] - mA[g], pa[(size_t)CC] - mA[g]);
      unsigned a23 = pk2bf(pa[(size_t)2 * CC] - mA[g], pa[(size_t)3 * CC] - mA[g]);
      unsigned b01 = pk2bf(pb[0] - mB[g], pb[(size_t)CC] - mB[g]);
      unsigned b23 = pk2bf(pb[(size_t)2 * CC] - mB[g], pb[(size_t)3 * CC] - mB[g]);
      *(unsigned long long*)&As[m][kg * 4] = a01 | ((unsigned long long)a23 << 32);
      *(unsigned long long*)&Bs[m][kg * 4] = b01 | ((unsigned long long)b23 << 32);
    }
    __syncthreads();

    v16bf a = mkfrag(*(const v4u*)&As[mrow][klo], *(const v4u*)&As[mrow][klo + 16]);
#pragma unroll
    for (int j = 0; j < NW; ++j) {
      int ncol = 16 * j + lcol;
      v16bf bb = mkfrag(*(const v4u*)&Bs[ncol][klo], *(const v4u*)&Bs[ncol][klo + 16]);
      acc8[j] = __builtin_amdgcn_wmma_f32_16x16x32_bf16(
          false, a, false, bb, (short)0, acc8[j], false, false);
    }
    __syncthreads();
  }

  const int half8 = (lane < 16) ? 0 : 8;
#pragma unroll
  for (int j = 0; j < NW; ++j) {
    int gj = j0 + 16 * j + lcol;
#pragma unroll
    for (int r = 0; r < 8; ++r) {
      int gi = i0 + 16 * w + r + half8;
      Gpart[((size_t)z * CC + gi) * CC + gj] = acc8[j][r];
    }
  }
}

// fixed-order slice sum -> off-diagonal sumsq -> acc
__global__ void vic_cov_reduce(const float* __restrict__ Gpart, int nsl, int n,
                               float* acc) {
  int idx = blockIdx.x * blockDim.x + threadIdx.x;   // 0 .. 768*768-1
  int gi = idx / CC, gj = idx % CC;
  float g = 0.f;
  for (int z = 0; z < nsl; ++z) g += Gpart[((size_t)z * CC + gi) * CC + gj];
  float v = g * (1.0f / (float)(n - 1));
  float sum = (gi != gj) ? v * v : 0.0f;
#pragma unroll
  for (int off = 16; off; off >>= 1) sum += __shfl_xor(sum, off, 32);
  if ((threadIdx.x & 31) == 0) atomicAdd(acc, sum * (1.0f / CC));
}

// ---------------------------------------------------------------- final combine
__global__ void vic_final(const float* s, float* out) {
  float invg = s[0] * (1.0f / (CB * CC));
  float stdg = (s[1] + s[2]) * (0.5f / CC);
  float covg = s[3] + s[4];
  float gl = 25.0f * invg + 25.0f * stdg + covg;

  float denom = 1.0f / (float)(CB * NM * CC);
  float inv1 = 25.0f * s[13] * denom;
  float inv2 = 25.0f * s[14] * denom;
  float var1 = 25.0f * (s[5] + s[6]) * (0.5f / CC);
  float var2 = 25.0f * (s[7] + s[8]) * (0.5f / CC);
  float cov1 = s[9] + s[10];
  float cov2 = s[11] + s[12];
  float ll = 0.5f * (inv1 + inv2) + 0.5f * (var1 + var2) + 0.5f * (cov1 + cov2);
  out[0] = 0.5f * gl + 0.5f * ll;
}

// ---------------------------------------------------------------- launch
extern "C" void kernel_launch(void* const* d_in, const int* in_sizes, int n_in,
                              void* d_out, int out_size, void* d_ws, size_t ws_size,
                              hipStream_t stream) {
  const float* S1 = (const float*)d_in[0];   // (64,768,24,24)
  const float* P1 = (const float*)d_in[1];   // (64,768)
  const float* S2 = (const float*)d_in[2];
  const float* P2 = (const float*)d_in[3];
  float* out = (float*)d_out;

  float* s      = (float*)d_ws;              // 32 scalar slots
  float* mean1  = s + 32;                    // 768
  float* mean2  = mean1 + CC;                // 768
  float* meanf  = mean2 + CC;                // 4*768
  float* mpart  = meanf + 4 * CC;            // 4*16*768
  float* m2part = mpart + 4 * NSL * CC;      // 4*16*768
  float* sq1    = m2part + 4 * NSL * CC;     // 64*576
  float* sq2    = sq1 + CB * HW;             // 64*576
  unsigned long long* nnmin = (unsigned long long*)(sq2 + CB * HW);  // 2*64*576 u64
  float* Gpart  = (float*)(nnmin + 2 * CB * HW);                     // CZ*768*768
  float* filt   = Gpart + (size_t)CZ * CC * CC;                      // 4*64*50*768

  vic_init<<<(2 * CB * HW) / 256, 256, 0, stream>>>(s, nnmin);
  vic_pooled_stats<<<CC / 256, 256, 0, stream>>>(P1, P2, mean1, mean2, s);
  vic_sqsum<<<(CB * HW) / 256, 256, 0, stream>>>(S1, S2, sq1, sq2);
  vic_dist<<<dim3(HW / TT, HW / TT, CB), NTH, 0, stream>>>(S1, S2, sq1, sq2,
                                                           nnmin, nnmin + CB * HW);
  vic_select<<<dim3(CB, 2), HW, 0, stream>>>(S1, S2, nnmin, filt, s);
  vic_filt_moments<<<dim3(CC / 256, 4, NSL), 256, 0, stream>>>(filt, mpart, m2part);
  vic_filt_fin<<<(4 * CC) / 256, 256, 0, stream>>>(mpart, m2part, meanf, s);

  vic_cov_part<<<dim3(CC / TT, CC / TT, 1), NTH, 0, stream>>>(P1, mean1, CB, Gpart);
  vic_cov_reduce<<<(CC * CC) / 256, 256, 0, stream>>>(Gpart, 1, CB, &s[3]);
  vic_cov_part<<<dim3(CC / TT, CC / TT, 1), NTH, 0, stream>>>(P2, mean2, CB, Gpart);
  vic_cov_reduce<<<(CC * CC) / 256, 256, 0, stream>>>(Gpart, 1, CB, &s[4]);
  for (int q = 0; q < 4; ++q) {
    vic_cov_part<<<dim3(CC / TT, CC / TT, CZ), NTH, 0, stream>>>(
        filt + (size_t)q * (CB * NM) * CC, meanf + q * CC, CB * NM, Gpart);
    vic_cov_reduce<<<(CC * CC) / 256, 256, 0, stream>>>(Gpart, CZ, CB * NM, &s[9 + q]);
  }
  vic_final<<<1, 1, 0, stream>>>(s, out);
}
